// MiniCPMV4_6ViTWindowAttentionMerger_57071525429573
// MI455X (gfx1250) — compile-verified, matrix operand order checked by reading, not run on hardware
//
#include <hip/hip_runtime.h>
#include <math.h>

// ---------------------------------------------------------------------------
// Types for CDNA5 WMMA (wave32, 16x16x32 bf16 -> f32 accum) and TDM
// ---------------------------------------------------------------------------
typedef __attribute__((ext_vector_type(16))) __bf16 v16bf;
typedef __attribute__((ext_vector_type(8)))  float  v8f;
typedef __attribute__((ext_vector_type(4))) unsigned int v4ui;
typedef __attribute__((ext_vector_type(8))) int v8i_t;
typedef __attribute__((ext_vector_type(4))) int v4i_t;

__device__ __forceinline__ unsigned short f2bf(float x) {
    unsigned u = __float_as_uint(x);
    u += 0x7FFFu + ((u >> 16) & 1u);   // round-to-nearest-even
    return (unsigned short)(u >> 16);
}
__device__ __forceinline__ float bf2f(unsigned short h) {
    return __uint_as_float(((unsigned)h) << 16);
}
__device__ __forceinline__ float gelu_tanh(float x) {
    float x3 = x * x * x;
    return 0.5f * x * (1.f + tanhf(0.7978845608028654f * (x + 0.044715f * x3)));
}

// ---------------------------------------------------------------------------
// Problem constants (uniform tgt_sizes==64 case, matching the reference)
// ---------------------------------------------------------------------------
#define BB   8
#define LL   4096          // 64*64
#define DD   1152
#define D4   4608
#define I4   17216
#define NWIN 8192          // B * 32 * 32
#define MTOK 32768         // B * L  (window-order token rows)
#define NH   16
#define HD   72

// GEMM tiling
#define BM 128
#define BN 128
#define BK 32
#define LDT 40             // padded LDS row stride (elements): 80B, still 16B aligned

// ---------------------------------------------------------------------------
// Tensor Data Mover: 2-D tile (tile_h rows x tile_w elems of 2B) -> LDS,
// padded to LDT-element rows via pad_interval=3 (64B) / pad_amount=3 (16B).
// Descriptor bit layout per CDNA5 ISA ch.8 (D# groups 0/1; groups 2/3 unused
// for 2-D tensors and passed as zeros). Rows beyond rows_avail zero-fill.
// ---------------------------------------------------------------------------
__device__ __forceinline__ void tdm_load_2d_bf16(
    unsigned lds_byte_addr,           // LDS destination byte address
    const unsigned short* gptr,       // global tile start
    unsigned rows_avail,              // tensor_dim1 (rows available; OOB -> 0)
    unsigned row_len_elts,            // tensor_dim0 (elements left in a row)
    unsigned row_stride_elts,         // tensor_dim0_stride (elements)
    unsigned tile_w, unsigned tile_h) // tile_dim0 / tile_dim1
{
    unsigned long long ga = (unsigned long long)(size_t)gptr;
    v4ui g0;
    g0[0] = 1u;                                        // count=1, user desc
    g0[1] = lds_byte_addr;                             // lds_addr
    g0[2] = (unsigned)(ga & 0xFFFFFFFFu);              // global_addr[31:0]
    g0[3] = (unsigned)((ga >> 32) & 0x01FFFFFFu)       // global_addr[56:32]
          | (2u << 30);                                // type = 2 ("image")
    v8i_t g1;
    g1[0] = (int)((1u << 16)                           // data_size = 1 (2B)
          | (1u << 20)                                 // pad_enable
          | (3u << 22)                                 // pad_interval: 16 DW = 64B
          | (3u << 25));                               // pad_amount:    4 DW = 16B
    g1[1] = (int)((row_len_elts & 0xFFFFu) << 16);     // tensor_dim0[15:0]
    g1[2] = (int)((row_len_elts >> 16)
          | ((rows_avail & 0xFFFFu) << 16));           // dim0[31:16] | dim1[15:0]
    g1[3] = (int)((rows_avail >> 16)
          | (tile_w << 16));                           // dim1[31:16] | tile_dim0
    g1[4] = (int)(tile_h & 0xFFFFu);                   // tile_dim1 (tile_dim2=0)
    g1[5] = (int)row_stride_elts;                      // tensor_dim0_stride[31:0]
    g1[6] = 0;                                         // stride[47:32] | dim1_stride lo
    g1[7] = 0;
    v4i_t gz = {0, 0, 0, 0};
#if defined(__clang_major__) && (__clang_major__ >= 23)
    v8i_t gz8 = {0, 0, 0, 0, 0, 0, 0, 0};
    __builtin_amdgcn_tensor_load_to_lds(g0, g1, gz, gz, gz8, 0);
#else
    __builtin_amdgcn_tensor_load_to_lds(g0, g1, gz, gz, 0);
#endif
}

// ---------------------------------------------------------------------------
// f32 -> bf16 weight conversion (grid-stride)
// ---------------------------------------------------------------------------
__global__ void cvt_f32_bf16(const float* __restrict__ src,
                             unsigned short* __restrict__ dst, long long n) {
    long long i = (long long)blockIdx.x * blockDim.x + threadIdx.x;
    long long stride = (long long)gridDim.x * blockDim.x;
    for (; i < n; i += stride) dst[i] = f2bf(src[i]);
}

// ---------------------------------------------------------------------------
// LayerNorm(D=1152) + to_win permutation, output bf16 in window order
// ---------------------------------------------------------------------------
__global__ void ln_to_win(const float* __restrict__ x,
                          const float* __restrict__ g,
                          const float* __restrict__ b,
                          unsigned short* __restrict__ out) {
    __shared__ float s1[256], s2[256];
    const int r = blockIdx.x;
    const int bI  = r >> 12;
    const int rem = r & 4095;
    const int win = rem >> 2, t = rem & 3;
    const int i = win >> 5, j = win & 31, p = t >> 1, q = t & 1;
    const int l = ((2 * i + p) << 6) + (2 * j + q);
    const float* xr = x + ((size_t)bI * LL + l) * DD;

    float s = 0.f, ss = 0.f;
    for (int d = threadIdx.x; d < DD; d += 256) {
        float v = xr[d]; s += v; ss += v * v;
    }
    s1[threadIdx.x] = s; s2[threadIdx.x] = ss;
    __syncthreads();
    for (int off = 128; off > 0; off >>= 1) {
        if (threadIdx.x < off) {
            s1[threadIdx.x] += s1[threadIdx.x + off];
            s2[threadIdx.x] += s2[threadIdx.x + off];
        }
        __syncthreads();
    }
    const float m  = s1[0] * (1.f / DD);
    const float vv = s2[0] * (1.f / DD) - m * m;
    const float rs = rsqrtf(vv + 1e-6f);
    unsigned short* orow = out + (size_t)r * DD;
    for (int d = threadIdx.x; d < DD; d += 256)
        orow[d] = f2bf((xr[d] - m) * rs * g[d] + b[d]);
}

// ---------------------------------------------------------------------------
// TDM-fed, double-buffered bf16 WMMA GEMM:  C[M,N] = A[M,K] * B[N,K]^T
//   EPI 0: +bias            -> bf16 outB
//   EPI 1: gelu(+bias)      -> bf16 outB
//   EPI 2: +bias + windowed residual gather from extra (=hidden)  -> f32 outF
//   EPI 3: +bias + extra[row*N+col] (=res2)                       -> f32 outF
// 256 threads = 8 waves (2x4); wave owns 64x32 of C (4x2 WMMA tiles).
// Wave 0 drives the Tensor Data Mover; N-edge handled by D# OOB zero-fill.
// ---------------------------------------------------------------------------
template <int EPI>
__global__ __launch_bounds__(256, 2)
void gemm_bf16_wmma(const unsigned short* __restrict__ A,
                    const unsigned short* __restrict__ B,
                    const float* __restrict__ bias,
                    int M, int N, int K,
                    unsigned short* __restrict__ outB,
                    float* __restrict__ outF,
                    const float* __restrict__ extra) {
    __shared__ __align__(16) unsigned short As[2][BM * LDT];
    __shared__ __align__(16) unsigned short Bs[2][BN * LDT];

    const int tid  = threadIdx.x;
    const int lane = tid & 31;
    const int wave = tid >> 5;
    const int wr   = wave >> 2;          // 0..1
    const int wc   = wave & 3;           // 0..3
    const int tileM = blockIdx.y * BM;
    const int tileN = blockIdx.x * BN;

    v8f acc[4][2];
    {
        v8f z = {0.f, 0.f, 0.f, 0.f, 0.f, 0.f, 0.f, 0.f};
        for (int mi = 0; mi < 4; ++mi)
            for (int ni = 0; ni < 2; ++ni) acc[mi][ni] = z;
    }

    const int khalf = (lane >> 4) * 8;   // fragment k base per ISA layout
    const int lm = lane & 15;
    const unsigned rowsB = (unsigned)(N - tileN);   // D# OOB zero-fills beyond

    union Frag { uint4 u[2]; v16bf v; };

    // ---- prologue: DMA first tile pair into buffer 0 ----
    if (wave == 0) {
        tdm_load_2d_bf16((unsigned)(size_t)&As[0][0],
                         A + (size_t)tileM * K, BM, (unsigned)K, (unsigned)K, BK, BM);
        tdm_load_2d_bf16((unsigned)(size_t)&Bs[0][0],
                         B + (size_t)tileN * K, rowsB, (unsigned)K, (unsigned)K, BK, BN);
    }

    int buf = 0;
    for (int k0 = 0; k0 < K; k0 += BK) {
        // current buffer's DMA complete -> publish to all waves
        if (wave == 0) __builtin_amdgcn_s_wait_tensorcnt(0);
        __syncthreads();

        // kick off next tile pair into the other buffer while we compute
        if (k0 + BK < K && wave == 0) {
            const int kn = k0 + BK;
            tdm_load_2d_bf16((unsigned)(size_t)&As[buf ^ 1][0],
                             A + (size_t)tileM * K + kn, BM,
                             (unsigned)(K - kn), (unsigned)K, BK, BM);
            tdm_load_2d_bf16((unsigned)(size_t)&Bs[buf ^ 1][0],
                             B + (size_t)tileN * K + kn, rowsB,
                             (unsigned)(K - kn), (unsigned)K, BK, BN);
        }

        // ---- LDS -> fragments (ISA 16-bit A/B layout: two b128 per lane) ----
        const unsigned short* as = &As[buf][0];
        const unsigned short* bs = &Bs[buf][0];
        Frag afr[4], bfr[2];
        for (int mi = 0; mi < 4; ++mi) {
            const int r = wr * 64 + mi * 16 + lm;
            afr[mi].u[0] = *(const uint4*)&as[r * LDT + khalf];
            afr[mi].u[1] = *(const uint4*)&as[r * LDT + 16 + khalf];
        }
        for (int ni = 0; ni < 2; ++ni) {
            const int r = wc * 32 + ni * 16 + lm;
            bfr[ni].u[0] = *(const uint4*)&bs[r * LDT + khalf];
            bfr[ni].u[1] = *(const uint4*)&bs[r * LDT + 16 + khalf];
        }

        // ---- 8 WMMAs: v_wmma_f32_16x16x32_bf16 ----
        for (int mi = 0; mi < 4; ++mi)
            for (int ni = 0; ni < 2; ++ni)
                acc[mi][ni] = __builtin_amdgcn_wmma_f32_16x16x32_bf16(
                    false, afr[mi].v, false, bfr[ni].v,
                    (short)0, acc[mi][ni], false, false);

        buf ^= 1;
    }

    // ---- epilogue: C/D layout -> row = v + 8*(lane>=16), col = lane&15 ----
    for (int mi = 0; mi < 4; ++mi) {
        for (int ni = 0; ni < 2; ++ni) {
            const int col = tileN + wc * 32 + ni * 16 + lm;
            if (col >= N) continue;
            const float bv = bias ? bias[col] : 0.f;
            const v8f c = acc[mi][ni];
            const int rbase = tileM + wr * 64 + mi * 16 + ((lane >> 4) << 3);
            for (int v = 0; v < 8; ++v) {
                const int row = rbase + v;
                float x = c[v] + bv;
                const size_t idx = (size_t)row * N + col;
                if (EPI == 0) {
                    outB[idx] = f2bf(x);
                } else if (EPI == 1) {
                    outB[idx] = f2bf(gelu_tanh(x));
                } else if (EPI == 2) {
                    // windowed residual gather: row -> (b,l)
                    const int bI  = row >> 12;
                    const int rem = row & 4095;
                    const int win = rem >> 2, t = rem & 3;
                    const int i = win >> 5, j = win & 31;
                    const int p = t >> 1,  q = t & 1;
                    const int l = ((2 * i + p) << 6) + (2 * j + q);
                    outF[idx] = x + extra[((size_t)bI * LL + l) * DD + col];
                } else {
                    outF[idx] = x + extra[idx];
                }
            }
        }
    }
}

// ---------------------------------------------------------------------------
// Tiny 4-token/window attention: softmax over 4 logits per (window, head)
// block = 64 threads: thread -> (head = tid/4, query = tid%4)
// ---------------------------------------------------------------------------
__global__ void win_attn(const unsigned short* __restrict__ q,
                         const unsigned short* __restrict__ k,
                         const unsigned short* __restrict__ v,
                         unsigned short* __restrict__ o) {
    const int w   = blockIdx.x;
    const int tid = threadIdx.x;
    const int h = tid >> 2, qi = tid & 3;
    const size_t base = (size_t)w * 4 * DD + (size_t)h * HD;
    const float scale = 0.11785113019775793f;   // 1/sqrt(72)

    float lg[4];
    for (int t = 0; t < 4; ++t) {
        float s = 0.f;
        const size_t qb = base + (size_t)qi * DD;
        const size_t kb = base + (size_t)t * DD;
        for (int d = 0; d < HD; ++d) s += bf2f(q[qb + d]) * bf2f(k[kb + d]);
        lg[t] = s * scale;
    }
    float mx = fmaxf(fmaxf(lg[0], lg[1]), fmaxf(lg[2], lg[3]));
    float e[4], sum = 0.f;
    for (int t = 0; t < 4; ++t) { e[t] = __expf(lg[t] - mx); sum += e[t]; }
    const float inv = 1.f / sum;
    for (int t = 0; t < 4; ++t) e[t] *= inv;

    const size_t ob = base + (size_t)qi * DD;
    for (int d = 0; d < HD; ++d) {
        float acc = 0.f;
        for (int t = 0; t < 4; ++t) acc += e[t] * bf2f(v[base + (size_t)t * DD + d]);
        o[ob + d] = f2bf(acc);
    }
}

// ---------------------------------------------------------------------------
// Merge prep per window: LN over flattened 4*D = 4608 (h2 is window order so
// the 4 rows are contiguous) -> xf bf16; res2 = mean of the 4 rows (f32)
// ---------------------------------------------------------------------------
__global__ void merge_prep(const float* __restrict__ h2,
                           const float* __restrict__ g,
                           const float* __restrict__ b,
                           unsigned short* __restrict__ xf,
                           float* __restrict__ res2) {
    __shared__ float s1[256], s2[256];
    const int w = blockIdx.x;
    const float* xr = h2 + (size_t)w * D4;

    float s = 0.f, ss = 0.f;
    for (int d = threadIdx.x; d < D4; d += 256) {
        float vv = xr[d]; s += vv; ss += vv * vv;
    }
    s1[threadIdx.x] = s; s2[threadIdx.x] = ss;
    __syncthreads();
    for (int off = 128; off > 0; off >>= 1) {
        if (threadIdx.x < off) {
            s1[threadIdx.x] += s1[threadIdx.x + off];
            s2[threadIdx.x] += s2[threadIdx.x + off];
        }
        __syncthreads();
    }
    const float m  = s1[0] * (1.f / D4);
    const float vv = s2[0] * (1.f / D4) - m * m;
    const float rs = rsqrtf(vv + 1e-6f);

    unsigned short* xo = xf + (size_t)w * D4;
    for (int d = threadIdx.x; d < D4; d += 256)
        xo[d] = f2bf((xr[d] - m) * rs * g[d] + b[d]);

    float* ro = res2 + (size_t)w * DD;
    for (int d = threadIdx.x; d < DD; d += 256)
        ro[d] = 0.25f * (xr[d] + xr[DD + d] + xr[2 * DD + d] + xr[3 * DD + d]);
}

// ---------------------------------------------------------------------------
// Tail of d_out: new_tgt_sizes (16 int32 = 32) and zero attention mask (8192)
// ---------------------------------------------------------------------------
#define MERGED_ELTS (BB * 1024 * DD)   // 9,437,184
__global__ void write_tail(float* __restrict__ out) {
    const int i = blockIdx.x * blockDim.x + threadIdx.x;
    if (i < 16) ((int*)(out + MERGED_ELTS))[i] = 32;
    if (i < 8192) out[MERGED_ELTS + 16 + i] = 0.f;
}

// ---------------------------------------------------------------------------
// Host-side orchestration
// ---------------------------------------------------------------------------
extern "C" void kernel_launch(void* const* d_in, const int* in_sizes, int n_in,
                              void* d_out, int out_size, void* d_ws, size_t ws_size,
                              hipStream_t stream) {
    const float* hidden = (const float*)d_in[0];
    const float* Wq = (const float*)d_in[3];
    const float* bq = (const float*)d_in[4];
    const float* Wk = (const float*)d_in[5];
    const float* bk = (const float*)d_in[6];
    const float* Wv = (const float*)d_in[7];
    const float* bv = (const float*)d_in[8];
    const float* Wo = (const float*)d_in[9];
    const float* bo = (const float*)d_in[10];
    const float* ln1_g = (const float*)d_in[11];
    const float* ln1_b = (const float*)d_in[12];
    const float* pre_g = (const float*)d_in[13];
    const float* pre_b = (const float*)d_in[14];
    const float* W1 = (const float*)d_in[15];
    const float* b1 = (const float*)d_in[16];
    const float* W2 = (const float*)d_in[17];
    const float* b2 = (const float*)d_in[18];
    float* out = (float*)d_out;

    // ---- workspace layout (bytes, all 256B aligned) ----
    char* ws = (char*)d_ws;
    size_t off = 0;
    auto alloc = [&](size_t bytes) { char* p = ws + off; off += (bytes + 255) & ~(size_t)255; return p; };
    unsigned short* wq_b = (unsigned short*)alloc((size_t)DD * DD * 2);
    unsigned short* wk_b = (unsigned short*)alloc((size_t)DD * DD * 2);
    unsigned short* wv_b = (unsigned short*)alloc((size_t)DD * DD * 2);
    unsigned short* wo_b = (unsigned short*)alloc((size_t)DD * DD * 2);
    unsigned short* w1_b = (unsigned short*)alloc((size_t)I4 * D4 * 2);
    unsigned short* w2_b = (unsigned short*)alloc((size_t)DD * I4 * 2);
    unsigned short* wn   = (unsigned short*)alloc((size_t)MTOK * DD * 2);
    unsigned short* q_b  = (unsigned short*)alloc((size_t)MTOK * DD * 2);
    unsigned short* k_b  = (unsigned short*)alloc((size_t)MTOK * DD * 2);
    unsigned short* v_b  = (unsigned short*)alloc((size_t)MTOK * DD * 2);
    unsigned short* o_b  = (unsigned short*)alloc((size_t)MTOK * DD * 2);
    float*          h2   = (float*)         alloc((size_t)MTOK * DD * 4);
    unsigned short* xf   = (unsigned short*)alloc((size_t)NWIN * D4 * 2);
    float*          res2 = (float*)         alloc((size_t)NWIN * DD * 4);
    unsigned short* act1 = (unsigned short*)alloc((size_t)NWIN * I4 * 2);
    (void)ws_size; (void)n_in; (void)in_sizes; (void)out_size;

    // ---- 1) weights -> bf16 ----
    {
        dim3 blk(256);
        cvt_f32_bf16<<<dim3(4096), blk, 0, stream>>>(Wq, wq_b, (long long)DD * DD);
        cvt_f32_bf16<<<dim3(4096), blk, 0, stream>>>(Wk, wk_b, (long long)DD * DD);
        cvt_f32_bf16<<<dim3(4096), blk, 0, stream>>>(Wv, wv_b, (long long)DD * DD);
        cvt_f32_bf16<<<dim3(4096), blk, 0, stream>>>(Wo, wo_b, (long long)DD * DD);
        cvt_f32_bf16<<<dim3(16384), blk, 0, stream>>>(W1, w1_b, (long long)I4 * D4);
        cvt_f32_bf16<<<dim3(16384), blk, 0, stream>>>(W2, w2_b, (long long)DD * I4);
    }

    // ---- 2) LN1 + to_win ----
    ln_to_win<<<dim3(MTOK), dim3(256), 0, stream>>>(hidden, ln1_g, ln1_b, wn);

    // ---- 3) Q/K/V projections (M=32768, N=1152, K=1152) ----
    {
        dim3 grid(DD / BN, MTOK / BM), blk(256);
        gemm_bf16_wmma<0><<<grid, blk, 0, stream>>>(wn, wq_b, bq, MTOK, DD, DD, q_b, nullptr, nullptr);
        gemm_bf16_wmma<0><<<grid, blk, 0, stream>>>(wn, wk_b, bk, MTOK, DD, DD, k_b, nullptr, nullptr);
        gemm_bf16_wmma<0><<<grid, blk, 0, stream>>>(wn, wv_b, bv, MTOK, DD, DD, v_b, nullptr, nullptr);
    }

    // ---- 4) 4-token window attention ----
    win_attn<<<dim3(NWIN), dim3(64), 0, stream>>>(q_b, k_b, v_b, o_b);

    // ---- 5) O projection + windowed residual -> h2 (f32, window order) ----
    gemm_bf16_wmma<2><<<dim3(DD / BN, MTOK / BM), dim3(256), 0, stream>>>(
        o_b, wo_b, bo, MTOK, DD, DD, nullptr, h2, hidden);

    // ---- 6) merge prep: LN(4608) -> xf bf16, res2 = window mean ----
    merge_prep<<<dim3(NWIN), dim3(256), 0, stream>>>(h2, pre_g, pre_b, xf, res2);

    // ---- 7) fc1 + GELU (M=8192, N=17216, K=4608) ----
    gemm_bf16_wmma<1><<<dim3((I4 + BN - 1) / BN, NWIN / BM), dim3(256), 0, stream>>>(
        xf, w1_b, b1, NWIN, I4, D4, act1, nullptr, nullptr);

    // ---- 8) fc2 + res2 -> merged output (M=8192, N=1152, K=17216) ----
    gemm_bf16_wmma<3><<<dim3(DD / BN, NWIN / BM), dim3(256), 0, stream>>>(
        act1, w2_b, b2, NWIN, DD, I4, nullptr, out, res2);

    // ---- 9) tgt_sizes + mask tail ----
    write_tail<<<dim3(32), dim3(256), 0, stream>>>(out);
}
